// SBC_Net_80882824119011
// MI455X (gfx1250) — matrix operand order repaired and checked by reading, not compiled, not run on hardware
//
#include <hip/hip_runtime.h>
#include <cstdint>
#include <cstddef>

// Problem constants (match reference)
#define SEQ  4096
#define BB   8
#define DD   512
#define DFF_ 1024
#define NH   8
#define DK   64
#define NL   4

typedef __attribute__((ext_vector_type(16))) __bf16 v16bf;
typedef __attribute__((ext_vector_type(2)))  __bf16 v2bf;
typedef __attribute__((ext_vector_type(8)))  float  v8f;

union Frag {
    v16bf v;
    uint4 q[2];   // 32 bytes
};

__device__ __forceinline__ unsigned short f2bf(float x) {
    union { float f; unsigned int u; } c; c.f = x;
    unsigned int r = c.u + 0x7FFFu + ((c.u >> 16) & 1u);   // RNE
    return (unsigned short)(r >> 16);
}

// Pack two f32 -> two bf16 in one dword (v_cvt_pk_bf16_f32 when available).
__device__ __forceinline__ unsigned int pk2bf(float a, float b) {
#if __has_builtin(__builtin_amdgcn_cvt_pk_bf16_f32)
    union { v2bf v; unsigned int u; } c;
    c.v = __builtin_amdgcn_cvt_pk_bf16_f32(a, b);
    return c.u;
#else
    return (unsigned int)f2bf(a) | ((unsigned int)f2bf(b) << 16);
#endif
}

// ---------------------------------------------------------------------------
// Generic GEMM: C = act(A @ W + bias) (+ residual). A:(M,K) W:(K,N) row-major.
// M%128==0, N%128==0, K%32==0. ACT: 0=none, 1=relu, 2=elu+1.
// Block = 256 threads = 8 waves; block tile 128x128; wave tile 64x32.
// Double-buffered LDS: next K-tile global loads fly under current WMMAs.
// ---------------------------------------------------------------------------
template<int ACT>
__global__ __launch_bounds__(256) void gemm_bias_act(
    const float* __restrict__ A, const float* __restrict__ W,
    const float* __restrict__ bias, const float* __restrict__ res,
    float* __restrict__ C, int M, int N, int K)
{
    constexpr int LDT = 40;   // ushort stride: 80B rows, 16B-aligned, conflict-free
    __shared__ __align__(16) unsigned short Asub[2][128 * LDT];
    __shared__ __align__(16) unsigned short Bsub[2][128 * LDT];

    const int t    = threadIdx.x;
    const int lane = t & 31;
    const int wave = t >> 5;
    const int wm   = wave >> 2;        // 0..1  -> 64-row slab
    const int wn   = wave & 3;         // 0..3  -> 32-col slab
    const int m0   = blockIdx.x * 128;
    const int n0   = blockIdx.y * 128;
    const int l15  = lane & 15;
    const int lh   = lane >> 4;

    v8f acc[4][2] = {};

    // Per-thread tile coordinates (fixed across K-steps).
    const int arow[4] = { (t + 0)   >> 3, (t + 256) >> 3, (t + 512) >> 3, (t + 768) >> 3 };
    const int akq     = (t & 7) << 2;
    const int bkk[4]  = { (t + 0)   >> 5, (t + 256) >> 5, (t + 512) >> 5, (t + 768) >> 5 };
    const int bn      = (t & 31) << 2;

    float4 aReg[4], bReg[4];
    auto load_tile = [&](int k0) {
        #pragma unroll
        for (int i = 0; i < 4; ++i) {
            aReg[i] = *(const float4*)&A[(size_t)(m0 + arow[i]) * K + k0 + akq];
            bReg[i] = *(const float4*)&W[(size_t)(k0 + bkk[i]) * N + n0 + bn];
        }
    };
    auto store_tile = [&](int buf) {
        #pragma unroll
        for (int i = 0; i < 4; ++i) {
            *(uint2*)&Asub[buf][arow[i] * LDT + akq] =
                make_uint2(pk2bf(aReg[i].x, aReg[i].y), pk2bf(aReg[i].z, aReg[i].w));
            Bsub[buf][(bn + 0) * LDT + bkk[i]] = f2bf(bReg[i].x);
            Bsub[buf][(bn + 1) * LDT + bkk[i]] = f2bf(bReg[i].y);
            Bsub[buf][(bn + 2) * LDT + bkk[i]] = f2bf(bReg[i].z);
            Bsub[buf][(bn + 3) * LDT + bkk[i]] = f2bf(bReg[i].w);
        }
    };

    const int nk = K >> 5;
    load_tile(0);
    store_tile(0);
    __syncthreads();

    for (int kt = 0; kt < nk; ++kt) {
        const int cur = kt & 1;
        if (kt + 1 < nk) {
            load_tile((kt + 1) << 5);          // in flight during WMMAs below
            if (kt + 2 < nk) {                 // global_prefetch_b8 two tiles ahead
                __builtin_prefetch(&A[(size_t)(m0 + (t >> 1)) * K + ((kt + 2) << 5) + ((t & 1) << 4)], 0, 0);
                __builtin_prefetch(&W[(size_t)(((kt + 2) << 5) + (t >> 3)) * N + n0 + ((t & 7) << 4)], 0, 0);
            }
        }

        Frag aF[4], bF[2];
        #pragma unroll
        for (int mi = 0; mi < 4; ++mi) {    // A frag: lanes 0-15 M, half selects K-subblock
            int ar = (wm * 64 + mi * 16 + l15) * LDT;
            aF[mi].q[0] = *(const uint4*)&Asub[cur][ar + lh * 8];
            aF[mi].q[1] = *(const uint4*)&Asub[cur][ar + 16 + lh * 8];
        }
        #pragma unroll
        for (int ni = 0; ni < 2; ++ni) {    // B frag: lane=N, half selects K-subblock
            int br = (wn * 32 + ni * 16 + l15) * LDT + lh * 8;
            bF[ni].q[0] = *(const uint4*)&Bsub[cur][br];
            bF[ni].q[1] = *(const uint4*)&Bsub[cur][br + 16];
        }
        #pragma unroll
        for (int mi = 0; mi < 4; ++mi)
            #pragma unroll
            for (int ni = 0; ni < 2; ++ni)
                acc[mi][ni] = __builtin_amdgcn_wmma_f32_16x16x32_bf16(
                    false, aF[mi].v, false, bF[ni].v, (short)0, acc[mi][ni], false, false);

        if (kt + 1 < nk) {
            store_tile(1 - cur);
            __syncthreads();
        }
    }

    // ---- epilogue: bias + activation + residual ----
    #pragma unroll
    for (int ni = 0; ni < 2; ++ni) {
        const int col = n0 + wn * 32 + ni * 16 + l15;
        const float bv = bias[col];
        #pragma unroll
        for (int mi = 0; mi < 4; ++mi) {
            #pragma unroll
            for (int r = 0; r < 8; ++r) {
                const int row = m0 + wm * 64 + mi * 16 + r + (lh << 3);
                float c = acc[mi][ni][r] + bv;
                if (ACT == 1) c = fmaxf(c, 0.0f);
                else if (ACT == 2) c = (c > 0.0f) ? (c + 1.0f) : __expf(c);  // elu(x)+1
                if (res) c += res[(size_t)row * N + col];
                C[(size_t)row * N + col] = c;
            }
        }
    }
}

// ---------------------------------------------------------------------------
// context[b,h] = sum_s phiK[s]^T outer V[s]  (64x64, K=4096), + ksum column.
// ctx layout per (b,h): 64 x 80 f32.  cols 0..63 = context, col 64 = ksum, 65..79 = 0.
// One block per (b,h); each wave owns 2 of the 16 output 16x16 tiles.
// ---------------------------------------------------------------------------
__global__ __launch_bounds__(256) void linattn_context(
    const float* __restrict__ phiK, const float* __restrict__ V,
    float* __restrict__ ctx)
{
    constexpr int LDT = 136;  // 272B rows: 16B aligned, conflict-free frag reads
    __shared__ __align__(16) unsigned short KT[64 * LDT];  // [d][s]
    __shared__ __align__(16) unsigned short VT[64 * LDT];  // [e][s]
    __shared__ float ksumP[4][64];

    const int t    = threadIdx.x;
    const int lane = t & 31;
    const int wave = t >> 5;
    const int l15  = lane & 15, lh = lane >> 4;
    const int bh   = blockIdx.x;
    const int b    = bh >> 3, h = bh & 7;
    const size_t base = (size_t)b * DD + (size_t)h * DK;

    const int tile0 = wave * 2;
    v8f acc[2] = {};
    float ksum_acc = 0.0f;
    const int dloc = t & 63;
    const int sgrp = t >> 6;

    for (int sc = 0; sc < SEQ / 128; ++sc) {
        const int s0 = sc * 128;
        #pragma unroll 4
        for (int i = 0; i < 32; ++i) {
            int f  = t + i * 256;
            int si = f >> 6;        // sgrp + 4*i
            int d  = f & 63;        // == dloc
            size_t g = (size_t)(s0 + si) * (BB * DD) + base + d;
            float kv = phiK[g];
            float vv = V[g];
            KT[d * LDT + si] = f2bf(kv);
            VT[d * LDT + si] = f2bf(vv);
            ksum_acc += kv;
        }
        __syncthreads();
        #pragma unroll
        for (int w2 = 0; w2 < 2; ++w2) {
            const int tile = tile0 + w2;
            const int tm = tile >> 2, tn = tile & 3;
            #pragma unroll
            for (int ks = 0; ks < 4; ++ks) {
                Frag aF, bF;
                int ar = (tm * 16 + l15) * LDT + ks * 32;
                aF.q[0] = *(const uint4*)&KT[ar + lh * 8];
                aF.q[1] = *(const uint4*)&KT[ar + 16 + lh * 8];
                int br = (tn * 16 + l15) * LDT + ks * 32 + lh * 8;
                bF.q[0] = *(const uint4*)&VT[br];
                bF.q[1] = *(const uint4*)&VT[br + 16];
                acc[w2] = __builtin_amdgcn_wmma_f32_16x16x32_bf16(
                    false, aF.v, false, bF.v, (short)0, acc[w2], false, false);
            }
        }
        __syncthreads();
    }

    float* cb = ctx + (size_t)bh * (64 * 80);
    #pragma unroll
    for (int w2 = 0; w2 < 2; ++w2) {
        const int tile = tile0 + w2;
        const int tm = tile >> 2, tn = tile & 3;
        #pragma unroll
        for (int r = 0; r < 8; ++r) {
            int d = tm * 16 + r + (lh << 3);
            int e = tn * 16 + l15;
            cb[d * 80 + e] = acc[w2][r];
        }
    }
    ksumP[sgrp][dloc] = ksum_acc;
    __syncthreads();
    if (t < 64) {
        float ks = ksumP[0][t] + ksumP[1][t] + ksumP[2][t] + ksumP[3][t];
        cb[t * 80 + 64] = ks;
        #pragma unroll
        for (int j = 65; j < 80; ++j) cb[t * 80 + j] = 0.0f;
    }
}

// ---------------------------------------------------------------------------
// out[s,e] = (phiQ[s,:] @ ctx[:,0:64]) / (phiQ[s,:] @ ksum + eps), per (b,h).
// Block: 128 s-rows x 80 cols (5 N-tiles); wave owns 16 rows.
// ---------------------------------------------------------------------------
__global__ __launch_bounds__(256) void linattn_apply(
    const float* __restrict__ phiQ, const float* __restrict__ ctx,
    float* __restrict__ attnOut)
{
    constexpr int LQ = 72;   // 144B rows: 16B aligned, conflict-free
    __shared__ __align__(16) unsigned short Qs[128 * LQ];  // [s][d]
    __shared__ __align__(16) unsigned short CT[80 * LQ];   // [n][d]
    __shared__ float denom[128];

    const int t    = threadIdx.x;
    const int lane = t & 31, wave = t >> 5;
    const int l15  = lane & 15, lh = lane >> 4;
    const int s0   = blockIdx.x * 128;
    const int bh   = blockIdx.y;
    const int b    = bh >> 3, h = bh & 7;
    const size_t base = (size_t)b * DD + (size_t)h * DK;

    #pragma unroll 4
    for (int i = 0; i < 32; ++i) {
        int f  = t + i * 256;
        int si = f >> 6, d = f & 63;
        Qs[si * LQ + d] = f2bf(phiQ[(size_t)(s0 + si) * (BB * DD) + base + d]);
    }
    const float* cb = ctx + (size_t)bh * (64 * 80);
    #pragma unroll 4
    for (int i = 0; i < 20; ++i) {              // 5120 = 20*256
        int f = t + i * 256;
        int d = f / 80, n = f % 80;
        CT[n * LQ + d] = f2bf(cb[f]);
    }
    __syncthreads();

    v8f acc[5] = {};
    Frag aF[2];
    const int mrow = wave * 16 + l15;
    #pragma unroll
    for (int ks = 0; ks < 2; ++ks) {
        int ar = mrow * LQ + ks * 32;
        aF[ks].q[0] = *(const uint4*)&Qs[ar + lh * 8];
        aF[ks].q[1] = *(const uint4*)&Qs[ar + 16 + lh * 8];
    }
    #pragma unroll
    for (int tn = 0; tn < 5; ++tn) {
        #pragma unroll
        for (int ks = 0; ks < 2; ++ks) {
            Frag bF;
            int br = (tn * 16 + l15) * LQ + ks * 32 + lh * 8;
            bF.q[0] = *(const uint4*)&CT[br];
            bF.q[1] = *(const uint4*)&CT[br + 16];
            acc[tn] = __builtin_amdgcn_wmma_f32_16x16x32_bf16(
                false, aF[ks].v, false, bF.v, (short)0, acc[tn], false, false);
        }
    }
    if (l15 == 0) {                              // tile 4, column 0 = denom
        #pragma unroll
        for (int r = 0; r < 8; ++r)
            denom[wave * 16 + r + (lh << 3)] = acc[4][r];
    }
    __syncthreads();
    #pragma unroll
    for (int tn = 0; tn < 4; ++tn) {
        #pragma unroll
        for (int r = 0; r < 8; ++r) {
            int sl = wave * 16 + r + (lh << 3);
            float val = acc[tn][r] / (denom[sl] + 1e-6f);
            attnOut[(size_t)(s0 + sl) * (BB * DD) + base + tn * 16 + l15] = val;
        }
    }
}

// ---------------------------------------------------------------------------
// Row-wise LayerNorm over D=512. One block (256 thr) per row.
// ---------------------------------------------------------------------------
__global__ __launch_bounds__(256) void layernorm_k(
    const float* __restrict__ x, const float* __restrict__ g,
    const float* __restrict__ bta, float* __restrict__ y)
{
    const int row = blockIdx.x;
    const int t   = threadIdx.x;
    const float* xr = x + (size_t)row * DD;
    float v0 = xr[t], v1 = xr[t + 256];
    float s1 = v0 + v1;
    float s2 = v0 * v0 + v1 * v1;
    #pragma unroll
    for (int off = 16; off > 0; off >>= 1) {
        s1 += __shfl_down(s1, off, 32);
        s2 += __shfl_down(s2, off, 32);
    }
    __shared__ float r1[8], r2[8], stats[2];
    if ((t & 31) == 0) { r1[t >> 5] = s1; r2[t >> 5] = s2; }
    __syncthreads();
    if (t == 0) {
        float a = 0.0f, c = 0.0f;
        #pragma unroll
        for (int i = 0; i < 8; ++i) { a += r1[i]; c += r2[i]; }
        float mean = a * (1.0f / DD);
        float var  = c * (1.0f / DD) - mean * mean;
        stats[0] = mean;
        stats[1] = rsqrtf(var + 1e-5f);
    }
    __syncthreads();
    float mean = stats[0], inv = stats[1];
    y[(size_t)row * DD + t]       = (v0 - mean) * inv * g[t]       + bta[t];
    y[(size_t)row * DD + t + 256] = (v1 - mean) * inv * g[t + 256] + bta[t + 256];
}

// ---------------------------------------------------------------------------
extern "C" void kernel_launch(void* const* d_in, const int* in_sizes, int n_in,
                              void* d_out, int out_size, void* d_ws, size_t ws_size,
                              hipStream_t stream) {
    (void)in_sizes; (void)n_in; (void)out_size; (void)ws_size;
    const float* src = (const float*)d_in[0];
    const float* Wq  = (const float*)d_in[1];
    const float* bq  = (const float*)d_in[2];
    const float* Wk  = (const float*)d_in[3];
    const float* bk  = (const float*)d_in[4];
    const float* Wv  = (const float*)d_in[5];
    const float* bv  = (const float*)d_in[6];
    const float* Wo  = (const float*)d_in[7];
    const float* bo  = (const float*)d_in[8];
    const float* g1  = (const float*)d_in[9];
    const float* be1 = (const float*)d_in[10];
    const float* W1  = (const float*)d_in[11];
    const float* b1  = (const float*)d_in[12];
    const float* W2  = (const float*)d_in[13];
    const float* b2  = (const float*)d_in[14];
    const float* g2  = (const float*)d_in[15];
    const float* be2 = (const float*)d_in[16];
    float* out = (float*)d_out;

    const int    MROWS = SEQ * BB;              // 32768
    const size_t TSZ   = (size_t)MROWS * DD;    // 16.78M floats
    // ws layout: X | phiQ | phiK | V | ctx   (~270 MB). FFN hidden reuses K+V.
    float* X    = (float*)d_ws;
    float* bufQ = X + TSZ;
    float* bufK = bufQ + TSZ;
    float* bufV = bufK + TSZ;
    float* ctx  = bufV + TSZ;
    float* hbuf = bufK;                          // 2*TSZ = 32768 x 1024

    const dim3 blk(256);
    const dim3 gN512(MROWS / 128, DD / 128);
    const dim3 gN1024(MROWS / 128, DFF_ / 128);
    const dim3 gCtx(BB * NH);
    const dim3 gApp(SEQ / 128, BB * NH);
    const dim3 gLN(MROWS);

    for (int i = 0; i < NL; ++i) {
        const float* xin = (i == 0) ? src : X;
        const size_t wO = (size_t)i * DD * DD;
        const size_t w1 = (size_t)i * DD * DFF_;
        const size_t w2 = (size_t)i * DFF_ * DD;
        // QKV (phi = elu+1 fused into Q,K epilogues)
        gemm_bias_act<2><<<gN512, blk, 0, stream>>>(xin, Wq + wO, bq + i * DD, nullptr, bufQ, MROWS, DD, DD);
        gemm_bias_act<2><<<gN512, blk, 0, stream>>>(xin, Wk + wO, bk + i * DD, nullptr, bufK, MROWS, DD, DD);
        gemm_bias_act<0><<<gN512, blk, 0, stream>>>(xin, Wv + wO, bv + i * DD, nullptr, bufV, MROWS, DD, DD);
        // linear attention
        linattn_context<<<gCtx, blk, 0, stream>>>(bufK, bufV, ctx);
        linattn_apply<<<gApp, blk, 0, stream>>>(bufQ, ctx, bufV);     // attn -> bufV
        // output projection + residual, then LN1
        gemm_bias_act<0><<<gN512, blk, 0, stream>>>(bufV, Wo + wO, bo + i * DD, xin, bufK, MROWS, DD, DD);
        layernorm_k<<<gLN, blk, 0, stream>>>(bufK, g1 + i * DD, be1 + i * DD, X);
        // FFN + residual, then LN2
        gemm_bias_act<1><<<gN1024, blk, 0, stream>>>(X, W1 + w1, b1 + i * DFF_, nullptr, hbuf, MROWS, DFF_, DD);
        gemm_bias_act<0><<<gN512, blk, 0, stream>>>(hbuf, W2 + w2, b2 + i * DD, X, bufQ, MROWS, DD, DFF_);
        layernorm_k<<<gLN, blk, 0, stream>>>(bufQ, g2 + i * DD, be2 + i * DD, (i == NL - 1) ? out : X);
    }
}
